// TransformerBlock_18313740550638
// MI455X (gfx1250) — matrix operand order repaired and checked by reading, not compiled, not run on hardware
//
// CDNA5 (gfx1250 / MI455X) transformer block: MHA + MoE, bf16 WMMA throughout.
// v3: 128x128 GEMM workgroup tile (8 WMMA/wave/K-step), double-buffered LDS
// with async global->LDS A-staging overlapped with WMMA compute.
#include <hip/hip_runtime.h>
#include <math.h>

typedef __attribute__((ext_vector_type(16))) __bf16 v16bf_t;
typedef __attribute__((ext_vector_type(8)))  __bf16 bf16x8_t;
typedef __attribute__((ext_vector_type(8)))  float  v8f_t;
typedef __attribute__((ext_vector_type(4)))  float  f4_t;

#define B_   2
#define S_   2048
#define H_   1024
#define NH_  16
#define HD_  64
#define I_   1024
#define ER_  14
#define ES_  2
#define M_   (B_ * S_)   /* 4096 tokens */

enum { MODE_BF16 = 0, MODE_GELU_BF16 = 1, MODE_F32_RES = 2,
       MODE_ACC = 3, MODE_ACC_GATE = 4 };

// ---------------------------------------------------------------------------
// helpers
// ---------------------------------------------------------------------------
__device__ inline v8f_t wmma_bf16(v16bf_t a, v16bf_t b, v8f_t c) {
  // D = A(16x32 bf16) * B(32x16 bf16) + C(16x16 f32); fp32 accum, RNE.
  return __builtin_amdgcn_wmma_f32_16x16x32_bf16(false, a, false, b,
                                                 (short)0, c, false, false);
}

// Build a 16-element bf16 fragment from two contiguous 8-element (16B) chunks.
__device__ inline v16bf_t ld16_2(const __bf16* p0, const __bf16* p1) {
  bf16x8_t a = *(const bf16x8_t*)p0;
  bf16x8_t b = *(const bf16x8_t*)p1;
  v16bf_t r;
#pragma unroll
  for (int i = 0; i < 8; ++i) { r[i] = a[i]; r[i + 8] = b[i]; }
  return r;
}
__device__ inline v16bf_t ld16_c(const __bf16* p) { return ld16_2(p, p + 8); }

// Async global->LDS copy, 16B per lane. Tracked by ASYNCcnt (CDNA5 ISA).
// Generic pointers to __shared__ carry the LDS offset in their low 32 bits.
__device__ inline void async_cp16(void* lds_dst, const void* gsrc) {
  asm volatile("global_load_async_to_lds_b128 %0, %1, off"
               :: "v"((unsigned)(size_t)lds_dst), "v"(gsrc)
               : "memory");
}
__device__ inline void wait_async() {
  asm volatile("s_wait_asynccnt 0" ::: "memory");
}

// ---------------------------------------------------------------------------
// LayerNorm: one 256-thread block per token (H=1024 -> 4 elems/thread).
// ---------------------------------------------------------------------------
template <bool WRITE_F32>
__global__ __launch_bounds__(256) void ln_kernel(
    const float* __restrict__ x, const float* __restrict__ sc,
    const float* __restrict__ bi, __bf16* __restrict__ out_bf,
    float* __restrict__ out_f) {
  __shared__ float red[16];
  const int tok = blockIdx.x;
  const int tid = threadIdx.x;
  const int lane = tid & 31, wave = tid >> 5;
  const float* row = x + (size_t)tok * H_;
  float v[4], s = 0.f, s2 = 0.f;
#pragma unroll
  for (int j = 0; j < 4; ++j) {
    v[j] = row[tid + j * 256];
    s += v[j]; s2 += v[j] * v[j];
  }
#pragma unroll
  for (int off = 16; off > 0; off >>= 1) {
    s  += __shfl_xor(s,  off, 32);
    s2 += __shfl_xor(s2, off, 32);
  }
  if (lane == 0) { red[wave] = s; red[8 + wave] = s2; }
  __syncthreads();
  float ts = 0.f, ts2 = 0.f;
#pragma unroll
  for (int w = 0; w < 8; ++w) { ts += red[w]; ts2 += red[8 + w]; }
  const float mu = ts * (1.f / H_);
  const float var = ts2 * (1.f / H_) - mu * mu;
  const float rs = rsqrtf(var + 1e-5f);
#pragma unroll
  for (int j = 0; j < 4; ++j) {
    const int h = tid + j * 256;
    const float xn = (v[j] - mu) * rs * sc[h] + bi[h];
    out_bf[(size_t)tok * H_ + h] = (__bf16)xn;
    if constexpr (WRITE_F32) out_f[(size_t)tok * H_ + h] = xn;
  }
}

// ---------------------------------------------------------------------------
// Tiled GEMM: C[M,N] = A[M,K](bf16) x W[K,N](fp32 -> bf16 in LDS) + bias.
// Workgroup tile 128x128, 8 waves as 4x2, each wave a 32x64 block
// (2 A-frags x 4 B-frags = 8 WMMAs per K-step of 32).
// Double-buffered LDS: tile t+1's async A-copy + W global loads are issued
// before tile t's WMMAs; fp32->bf16 convert/scatter happens after the math;
// a single barrier+asynccnt-wait per iteration closes the pipeline.
// ---------------------------------------------------------------------------
template <int MODE>
__global__ __launch_bounds__(256) void gemm_kernel(
    const __bf16* __restrict__ A, const float* __restrict__ W,
    const float* __restrict__ bias, void* __restrict__ Cout,
    const float* __restrict__ residual, const float* __restrict__ scale,
    int scale_stride, int K, int N) {
  __shared__ __bf16 Asm[2][128][40];  // [buf][m][k], 80B rows (16B multiple)
  __shared__ __bf16 Bsm[2][128][40];  // [buf][n][k] (transposed)
  const int tid = threadIdx.x;
  const int lane = tid & 31, wave = tid >> 5;
  const int m0 = blockIdx.y * 128, n0 = blockIdx.x * 128;
  const int wm = wave >> 1, wn = wave & 1;  // 4x2 wave grid over 128x128

  // staging geometry
  const int ar = tid >> 2, ac8 = (tid & 3) * 8;      // A: 2 x 16B per thread
  const int kr = tid >> 3, nb = (tid & 7) * 16;      // B: 32 rows x 128 cols

  v8f_t acc[2][4] = {};
  f4_t wreg[4];

  auto stage_a = [&](int k0, int buf) {
    async_cp16(&Asm[buf][ar][ac8],      &A[(size_t)(m0 + ar) * K + k0 + ac8]);
    async_cp16(&Asm[buf][ar + 64][ac8], &A[(size_t)(m0 + ar + 64) * K + k0 + ac8]);
  };
  auto load_b = [&](int k0) {
    const float* wp = &W[(size_t)(k0 + kr) * N + n0 + nb];
#pragma unroll
    for (int q = 0; q < 4; ++q) wreg[q] = *(const f4_t*)(wp + 4 * q);
  };
  auto store_b = [&](int buf) {
#pragma unroll
    for (int q = 0; q < 4; ++q)
#pragma unroll
      for (int j = 0; j < 4; ++j)
        Bsm[buf][nb + q * 4 + j][kr] = (__bf16)wreg[q][j];
  };

  // prologue: tile 0
  stage_a(0, 0);
  load_b(0);
  store_b(0);
  wait_async();
  __syncthreads();

  const int T = K / 32;
  for (int t = 0; t < T; ++t) {
    const int cur = t & 1, nxt = cur ^ 1;
    if (t + 1 < T) {  // kick next tile's data movement before the math
      stage_a((t + 1) * 32, nxt);
      load_b((t + 1) * 32);
      const int kp = (t + 2 < T) ? (t + 2) * 32 : (T - 1) * 32;
      __builtin_prefetch(&W[(size_t)(kp + kr) * N + n0 + nb], 0, 3);
    }

    v16bf_t af[2], bfr[4];
#pragma unroll
    for (int i = 0; i < 2; ++i) {  // A frag: lane = M | K-half, contiguous K
      const int m = wm * 32 + i * 16 + (lane & 15);
      const int kb = (lane >> 4) * 16;
      af[i] = ld16_c(&Asm[cur][m][kb]);
    }
#pragma unroll
    for (int j = 0; j < 4; ++j) {  // B frag: lane = N | K-half, contiguous K
      const int n = wn * 64 + j * 16 + (lane & 15);
      const int kb = (lane >> 4) * 16;
      bfr[j] = ld16_c(&Bsm[cur][n][kb]);
    }
#pragma unroll
    for (int i = 0; i < 2; ++i)
#pragma unroll
      for (int j = 0; j < 4; ++j)
        acc[i][j] = wmma_bf16(af[i], bfr[j], acc[i][j]);

    if (t + 1 < T) store_b(nxt);  // convert+scatter after WMMAs are issued
    wait_async();
    __syncthreads();
  }

  // epilogue: C layout — vgpr r -> M = r + 8*(lane>=16), N = lane&15
#pragma unroll
  for (int i = 0; i < 2; ++i) {
#pragma unroll
    for (int j = 0; j < 4; ++j) {
      const int n = n0 + wn * 64 + j * 16 + (lane & 15);
      const float bn = bias[n];
#pragma unroll
      for (int r = 0; r < 8; ++r) {
        const int m = m0 + wm * 32 + i * 16 + r + ((lane >> 4) << 3);
        const size_t idx = (size_t)m * N + n;
        const float val = acc[i][j][r] + bn;
        if constexpr (MODE == MODE_BF16) {
          ((__bf16*)Cout)[idx] = (__bf16)val;
        } else if constexpr (MODE == MODE_GELU_BF16) {
          const float g = 0.5f * val * (1.f + erff(val * 0.70710678118f));
          ((__bf16*)Cout)[idx] = (__bf16)g;
        } else if constexpr (MODE == MODE_F32_RES) {
          ((float*)Cout)[idx] = val + residual[idx];
        } else if constexpr (MODE == MODE_ACC) {
          ((float*)Cout)[idx] += val;   // disjoint per thread; launches serial
        } else {  // MODE_ACC_GATE
          ((float*)Cout)[idx] += scale[(size_t)m * scale_stride] * val;
        }
      }
    }
  }
}

// ---------------------------------------------------------------------------
// Flash attention: block = (q-tile of 128) x head x batch; 8 waves, each wave
// owns 16 q rows. K staged [key][hd] (async), V staged transposed [hd][key].
// ---------------------------------------------------------------------------
__global__ __launch_bounds__(256) void attn_kernel(
    const __bf16* __restrict__ q, const __bf16* __restrict__ k,
    const __bf16* __restrict__ v, __bf16* __restrict__ ctx) {
  __shared__ __bf16 Kt[32][72];       // [key][hd], 144B rows
  __shared__ __bf16 Vt[64][40];       // [hd][key]
  __shared__ __bf16 Pb[8][16][40];    // per-wave P scratch
  const int tid = threadIdx.x, lane = tid & 31, wave = tid >> 5;
  const int b = blockIdx.z, h = blockIdx.y, q0 = blockIdx.x * 128;
  const float scl = 0.125f;  // 1/sqrt(HD)

  v16bf_t qa[2];
  {  // Q A-fragments (HD split into two K=32 chunks)
    const int qrow = q0 + wave * 16 + (lane & 15);
    const int kb = (lane >> 4) * 16;
    const __bf16* qp = q + ((size_t)((b * S_ + qrow) * NH_ + h)) * HD_;
#pragma unroll
    for (int c = 0; c < 2; ++c)
      qa[c] = ld16_c(qp + 32 * c + kb);
  }

  v8f_t o[4] = {};
  float rowm[8], rowl[8];
#pragma unroll
  for (int r = 0; r < 8; ++r) { rowm[r] = -1e30f; rowl[r] = 0.f; }

  for (int kt = 0; kt < S_ / 32; ++kt) {
    __syncthreads();
    {  // stage K (async, row-major) and V (transposed through VGPRs)
      const int key = tid >> 3, c8 = (tid & 7) * 8;
      const size_t base = ((size_t)((b * S_ + kt * 32 + key) * NH_ + h)) * HD_;
      async_cp16(&Kt[key][c8], k + base + c8);
      bf16x8_t vv = *(const bf16x8_t*)(v + base + c8);
#pragma unroll
      for (int j = 0; j < 8; ++j) Vt[c8 + j][key] = vv[j];
    }
    wait_async();
    __syncthreads();

    // scores S(16x32) = Q * K^T   (B frag: n=key, K-dim=hd)
    v8f_t s0 = {}, s1 = {};
#pragma unroll
    for (int c = 0; c < 2; ++c) {
      const int kb = (lane >> 4) * 16;
      v16bf_t b0 = ld16_c(&Kt[(lane & 15)][32 * c + kb]);
      v16bf_t b1 = ld16_c(&Kt[16 + (lane & 15)][32 * c + kb]);
      s0 = wmma_bf16(qa[c], b0, s0);
      s1 = wmma_bf16(qa[c], b1, s1);
    }

    // online softmax; row reductions stay inside each 16-lane half (wave32)
#pragma unroll
    for (int r = 0; r < 8; ++r) {
      float a0 = s0[r] * scl, a1 = s1[r] * scl;
      float mx = fmaxf(a0, a1);
#pragma unroll
      for (int off = 8; off > 0; off >>= 1) mx = fmaxf(mx, __shfl_xor(mx, off, 32));
      const float nm = fmaxf(rowm[r], mx);
      const float corr = __expf(rowm[r] - nm);
      const float p0 = __expf(a0 - nm), p1 = __expf(a1 - nm);
      float sum = p0 + p1;
#pragma unroll
      for (int off = 8; off > 0; off >>= 1) sum += __shfl_xor(sum, off, 32);
      rowl[r] = rowl[r] * corr + sum;
      rowm[r] = nm;
#pragma unroll
      for (int t = 0; t < 4; ++t) o[t][r] *= corr;
      const int m = r + ((lane >> 4) << 3);
      Pb[wave][m][lane & 15] = (__bf16)p0;
      Pb[wave][m][16 + (lane & 15)] = (__bf16)p1;
    }

    // O += P(16x32) * V(32x64); P re-laid C->A through per-wave LDS
    {
      const int m = lane & 15, kb = (lane >> 4) * 16;
      v16bf_t pa = ld16_c(&Pb[wave][m][kb]);
      const int kb2 = (lane >> 4) * 16;
#pragma unroll
      for (int t = 0; t < 4; ++t) {
        v16bf_t vb = ld16_c(&Vt[t * 16 + (lane & 15)][kb2]);
        o[t] = wmma_bf16(pa, vb, o[t]);
      }
    }
  }

#pragma unroll
  for (int r = 0; r < 8; ++r) {
    const float inv = 1.f / rowl[r];
    const int qrow = q0 + wave * 16 + r + ((lane >> 4) << 3);
    __bf16* cp = ctx + ((size_t)((b * S_ + qrow) * NH_ + h)) * HD_;
#pragma unroll
    for (int t = 0; t < 4; ++t)
      cp[t * 16 + (lane & 15)] = (__bf16)(o[t][r] * inv);
  }
}

// ---------------------------------------------------------------------------
// Router: one wave per token; softmax over 14 experts, top-2 gates.
// ---------------------------------------------------------------------------
__global__ __launch_bounds__(256) void router_kernel(
    const float* __restrict__ x2, const float* __restrict__ rw,
    const float* __restrict__ rb, float* __restrict__ gates) {
  const int lane = threadIdx.x & 31, wave = threadIdx.x >> 5;
  const int tok = blockIdx.x * 8 + wave;
  float acc[ER_] = {};
  for (int hh = lane; hh < H_; hh += 32) {
    const float xv = x2[(size_t)tok * H_ + hh];
#pragma unroll
    for (int e = 0; e < ER_; ++e) acc[e] += xv * rw[hh * ER_ + e];
  }
#pragma unroll
  for (int e = 0; e < ER_; ++e)
#pragma unroll
    for (int off = 16; off > 0; off >>= 1) acc[e] += __shfl_xor(acc[e], off, 32);
  float lg[ER_], mx = -1e30f;
#pragma unroll
  for (int e = 0; e < ER_; ++e) { lg[e] = acc[e] + rb[e]; mx = fmaxf(mx, lg[e]); }
  float se = 0.f;
#pragma unroll
  for (int e = 0; e < ER_; ++e) { lg[e] = __expf(lg[e] - mx); se += lg[e]; }
  const float inv = 1.f / se;
  int i1 = 0;
#pragma unroll
  for (int e = 1; e < ER_; ++e) if (lg[e] > lg[i1]) i1 = e;
  int i2 = (i1 == 0) ? 1 : 0;
#pragma unroll
  for (int e = 0; e < ER_; ++e) if (e != i1 && lg[e] > lg[i2]) i2 = e;
  if (lane < ER_) {
    const float g = (lane == i1 || lane == i2) ? lg[lane] * inv : 0.f;
    gates[(size_t)tok * ER_ + lane] = g;
  }
}

// d_out = x2 + attn_out   (base for MoE accumulation)
__global__ __launch_bounds__(256) void addinit_kernel(
    const float* __restrict__ a, const float* __restrict__ b,
    float* __restrict__ o, int n) {
  const int i = blockIdx.x * 256 + threadIdx.x;
  if (i < n) o[i] = a[i] + b[i];
}

// ---------------------------------------------------------------------------
// launch
// ---------------------------------------------------------------------------
extern "C" void kernel_launch(void* const* d_in, const int* in_sizes, int n_in,
                              void* d_out, int out_size, void* d_ws,
                              size_t ws_size, hipStream_t stream) {
  (void)in_sizes; (void)n_in; (void)out_size; (void)ws_size;
  const float* x    = (const float*)d_in[0];
  const float* l1s  = (const float*)d_in[1];
  const float* l1b  = (const float*)d_in[2];
  const float* l2s  = (const float*)d_in[3];
  const float* l2b  = (const float*)d_in[4];
  const float* wq   = (const float*)d_in[5];
  const float* bq   = (const float*)d_in[6];
  const float* wk   = (const float*)d_in[7];
  const float* bk   = (const float*)d_in[8];
  const float* wv   = (const float*)d_in[9];
  const float* bv   = (const float*)d_in[10];
  const float* wo   = (const float*)d_in[11];
  const float* bo   = (const float*)d_in[12];
  const float* rww  = (const float*)d_in[13];
  const float* rwb  = (const float*)d_in[14];
  const float* sw1  = (const float*)d_in[15];
  const float* sb1  = (const float*)d_in[16];
  const float* sw2  = (const float*)d_in[17];
  const float* sb2  = (const float*)d_in[18];
  const float* rw1  = (const float*)d_in[19];
  const float* rb1  = (const float*)d_in[20];
  const float* rw2  = (const float*)d_in[21];
  const float* rb2  = (const float*)d_in[22];

  size_t off = 0;
  auto take = [&](size_t bytes) {
    void* p = (char*)d_ws + off;
    off += (bytes + 255) & ~(size_t)255;
    return p;
  };
  __bf16* xn     = (__bf16*)take((size_t)M_ * H_ * 2);   // ln1 out (bf16)
  __bf16* qb     = (__bf16*)take((size_t)M_ * H_ * 2);
  __bf16* kb     = (__bf16*)take((size_t)M_ * H_ * 2);
  __bf16* vb     = (__bf16*)take((size_t)M_ * H_ * 2);
  __bf16* ctxb   = (__bf16*)take((size_t)M_ * H_ * 2);   // attention context
  float*  attn_o = (float*)take((size_t)M_ * H_ * 4);    // attn_out (fp32)
  float*  x2f    = (float*)take((size_t)M_ * H_ * 4);    // ln2 out fp32
  __bf16* x2b    = (__bf16*)take((size_t)M_ * H_ * 2);   // ln2 out bf16
  __bf16* hbuf   = (__bf16*)take((size_t)M_ * I_ * 2);   // expert hidden
  float*  gates  = (float*)take((size_t)M_ * ER_ * 4);
  float*  outf   = (float*)d_out;

  const dim3 blk(256);
  const dim3 gH(H_ / 128, M_ / 128);   // N=1024 -> 8, M=4096 -> 32
  const dim3 gI(I_ / 128, M_ / 128);
  const dim3 gA(S_ / 128, NH_, B_);

  // --- attention ---
  ln_kernel<false><<<M_, blk, 0, stream>>>(x, l1s, l1b, xn, nullptr);
  gemm_kernel<MODE_BF16><<<gH, blk, 0, stream>>>(xn, wq, bq, qb, nullptr, nullptr, 0, H_, H_);
  gemm_kernel<MODE_BF16><<<gH, blk, 0, stream>>>(xn, wk, bk, kb, nullptr, nullptr, 0, H_, H_);
  gemm_kernel<MODE_BF16><<<gH, blk, 0, stream>>>(xn, wv, bv, vb, nullptr, nullptr, 0, H_, H_);
  attn_kernel<<<gA, blk, 0, stream>>>(qb, kb, vb, ctxb);
  gemm_kernel<MODE_F32_RES><<<gH, blk, 0, stream>>>(ctxb, wo, bo, attn_o, x, nullptr, 0, H_, H_);

  // --- MoE ---
  ln_kernel<true><<<M_, blk, 0, stream>>>(attn_o, l2s, l2b, x2b, x2f);
  router_kernel<<<M_ / 8, blk, 0, stream>>>(x2f, rww, rwb, gates);
  addinit_kernel<<<(M_ * H_) / 256, blk, 0, stream>>>(x2f, attn_o, outf, M_ * H_);

  for (int e = 0; e < ES_; ++e) {  // shared experts (gate = 1)
    gemm_kernel<MODE_GELU_BF16><<<gI, blk, 0, stream>>>(
        x2b, sw1 + (size_t)e * H_ * I_, sb1 + e * I_, hbuf,
        nullptr, nullptr, 0, H_, I_);
    gemm_kernel<MODE_ACC><<<gH, blk, 0, stream>>>(
        hbuf, sw2 + (size_t)e * I_ * H_, sb2 + e * H_, outf,
        nullptr, nullptr, 0, I_, H_);
  }
  for (int e = 0; e < ER_; ++e) {  // routed experts, dense then gate-scaled
    gemm_kernel<MODE_GELU_BF16><<<gI, blk, 0, stream>>>(
        x2b, rw1 + (size_t)e * H_ * I_, rb1 + e * I_, hbuf,
        nullptr, nullptr, 0, H_, I_);
    gemm_kernel<MODE_ACC_GATE><<<gH, blk, 0, stream>>>(
        hbuf, rw2 + (size_t)e * I_ * H_, rb2 + e * H_, outf,
        nullptr, gates + e, ER_, I_, H_);
  }
}
